// LaneRelationalAttention_27496380629537
// MI455X (gfx1250) — compile-verified, matrix operand order checked by reading, not run on hardware
//
#include <hip/hip_runtime.h>
#include <hip/hip_bf16.h>

typedef __attribute__((ext_vector_type(16))) _Float16 v16h;
typedef __attribute__((ext_vector_type(8)))  float    v8f;
typedef _Float16 h16;

#define T_DIM 30
#define B_DIM 32
#define V_DIM 32
#define F_DIM 128
#define L_DIM 256
#define LF_DIM 64
#define H_DIM 8
#define DH_DIM 16
#define HIST_DIM 10

// A/B operand K-index mapping for V_WMMA_*_16X16X32 f16 (ISA 7.12.2):
// lane = l + 16*g holds (row/col)=l ; element j -> K = 8*g + (j&7) + (j>=8 ? 16 : 0)
__device__ __forceinline__ int kmap(int j, int g) {
    return 8 * g + (j & 7) + ((j & 8) << 1);
}

__device__ __forceinline__ v8f wmma_f16(v16h a, v16h b, v8f c) {
    return __builtin_amdgcn_wmma_f32_16x16x32_f16(
        /*neg_a=*/false, a, /*neg_b=*/false, b,
        /*c_mod=*/(short)0, c, /*reuse_a=*/false, /*reuse_b=*/false);
}

// fast, branch-free log-sigmoid using hw v_exp/v_log
__device__ __forceinline__ float logsigmoid_fast(float x) {
    float ax = __builtin_fabsf(x);
    float mn = fminf(x, 0.f);
    return mn - __logf(1.f + __expf(-ax));
}

// ---------------------------------------------------------------------------
// Kernel 1: lane projections.  K = lanes@Wk^T + bk ; Vlog = logsigmoid(lanes@Wv^T + bv)
// One wave per (b, Ltile(16), head) tile. 8 waves / block.
// Kf layout:   [B,H,L,DH]  f16  (contiguous along DH -> B-operand of QK^T)
// Vlog layout: [B,H,DH,L]  f16  (contiguous along L  -> B-operand of P@Vlog)
// ---------------------------------------------------------------------------
__global__ __launch_bounds__(256) void lane_proj_kernel(
    const float* __restrict__ lanes,
    const float* __restrict__ Wk, const float* __restrict__ bk,
    const float* __restrict__ Wv, const float* __restrict__ bv,
    h16* __restrict__ Kf, h16* __restrict__ VlogT) {
    int lane  = threadIdx.x & 31;
    int wg    = blockIdx.x * 8 + (threadIdx.x >> 5);  // 0..4095
    int ftile = wg & 7;          // head
    int mtile = (wg >> 3) & 15;  // L tile
    int b     = wg >> 7;
    int g     = lane >> 4;
    int lan15 = lane & 15;
    int lrow  = mtile * 16 + lan15;   // A row (L)
    int fcol  = ftile * 16 + lan15;   // B col (output feature)

    v8f ck = {}; v8f cv = {};
#pragma unroll
    for (int ks = 0; ks < 2; ++ks) {
        v16h a, bkm, bvm;
#pragma unroll
        for (int j = 0; j < 16; ++j) {
            int k = ks * 32 + kmap(j, g);               // LF index
            a[j]   = (h16)lanes[(b * L_DIM + lrow) * LF_DIM + k];
            bkm[j] = (h16)Wk[fcol * LF_DIM + k];
            bvm[j] = (h16)Wv[fcol * LF_DIM + k];
        }
        ck = wmma_f16(a, bkm, ck);
        cv = wmma_f16(a, bvm, cv);
    }
    float biask = bk[fcol];
    float biasv = bv[fcol];
    int dh = lan15;
#pragma unroll
    for (int e = 0; e < 8; ++e) {
        int m = e + 8 * g;
        int l = mtile * 16 + m;
        float xk = ck[e] + biask;
        float ls = logsigmoid_fast(cv[e] + biasv);
        Kf[((size_t)(b * H_DIM + ftile) * L_DIM + l) * DH_DIM + dh] = (h16)xk;
        // transposed: 8 consecutive halves per lane (e -> l contiguous)
        VlogT[((size_t)(b * H_DIM + ftile) * DH_DIM + dh) * L_DIM + l] = (h16)ls;
    }
}

// ---------------------------------------------------------------------------
// Kernel 2: vehicle projections. v = vehicles + initial_pos ;
// q = v@Wq^T + bq ; r = v@Wr^T + br   (1x1 conv == linear over F)
// Rows = T*B*V = 30720. One wave per (Mtile, head). Output [T,B,H,V,DH] f16.
// ---------------------------------------------------------------------------
__global__ __launch_bounds__(256) void veh_proj_kernel(
    const float* __restrict__ veh, const float* __restrict__ ipos,
    const float* __restrict__ Wq, const float* __restrict__ bq,
    const float* __restrict__ Wr, const float* __restrict__ br,
    h16* __restrict__ qf, h16* __restrict__ rf) {
    int lane  = threadIdx.x & 31;
    int wg    = blockIdx.x * 8 + (threadIdx.x >> 5);  // 0..15359
    int ftile = wg & 7;
    int mtile = wg >> 3;        // 0..1919
    int g     = lane >> 4;
    int lan15 = lane & 15;
    int arow  = mtile * 16 + lan15;
    int fcol  = ftile * 16 + lan15;

    v8f cq = {}; v8f cr = {};
#pragma unroll
    for (int ks = 0; ks < 4; ++ks) {
        v16h a, b0, b1;
#pragma unroll
        for (int j = 0; j < 16; ++j) {
            int k = ks * 32 + kmap(j, g);
            a[j]  = (h16)(veh[(size_t)arow * F_DIM + k] + ipos[(size_t)arow * F_DIM + k]);
            b0[j] = (h16)Wq[fcol * F_DIM + k];
            b1[j] = (h16)Wr[fcol * F_DIM + k];
        }
        cq = wmma_f16(a, b0, cq);
        cr = wmma_f16(a, b1, cr);
    }
    float biasq = bq[fcol], biasr = br[fcol];
    int dh = lan15;
    int h  = ftile;
#pragma unroll
    for (int e = 0; e < 8; ++e) {
        int m   = e + 8 * g;
        int row = mtile * 16 + m;       // flat (t,b,v)
        int t   = row >> 10;            // /(B*V)=1024
        int rem = row & 1023;
        int b   = rem >> 5;
        int vv  = rem & 31;
        size_t idx = ((((size_t)t * B_DIM + b) * H_DIM + h) * V_DIM + vv) * DH_DIM + dh;
        qf[idx] = (h16)(cq[e] + biasq);
        rf[idx] = (h16)(cr[e] + biasr);
    }
}

// ---------------------------------------------------------------------------
// Kernel 3: mask = any(mask_lanes, axis=0)   [HIST,B,L] bool(byte) -> [B,L] u32
// ---------------------------------------------------------------------------
__global__ void mask_kernel(const unsigned char* __restrict__ ml,
                            unsigned int* __restrict__ mask) {
    int i = blockIdx.x * blockDim.x + threadIdx.x;
    if (i < B_DIM * L_DIM) {
        unsigned int m = 0;
        for (int hh = 0; hh < HIST_DIM; ++hh) m |= ml[hh * B_DIM * L_DIM + i];
        mask[i] = m ? 1u : 0u;
    }
}

// ---------------------------------------------------------------------------
// Kernel 4: attention per (t,b,h). 2 waves/block, wave = V-tile (16 rows).
// scores = QK^T/4 (K-dim 16 zero-padded to 32), masked softmax over L=256,
// att = exp(P @ Vlog), rat = r * att  -> f16 [T,B,H,V,DH] (== reshape [T,B,V,F])
// ---------------------------------------------------------------------------
__global__ __launch_bounds__(64) void attn_kernel(
    const h16* __restrict__ qf, const h16* __restrict__ Kf,
    const h16* __restrict__ VlogT, const unsigned int* __restrict__ mask,
    const h16* __restrict__ rf, h16* __restrict__ rat) {
    __shared__ _Float16 pbuf[2][16][L_DIM + 8];
    int lane  = threadIdx.x & 31;
    int wv    = threadIdx.x >> 5;   // V tile 0/1
    int bid   = blockIdx.x;
    int h     = bid & 7;
    int b     = (bid >> 3) & 31;
    int t     = bid >> 8;
    int g     = lane >> 4;
    int lan15 = lane & 15;

    // Q tile A-operand, zero-padded K 16->32
    const h16* qbase = qf + (((size_t)t * B_DIM + b) * H_DIM + h) * (size_t)(V_DIM * DH_DIM);
    int qrow = wv * 16 + lan15;
    v16h aq;
#pragma unroll
    for (int j = 0; j < 16; ++j) {
        int k = kmap(j, g);
        aq[j] = (k < DH_DIM) ? qbase[qrow * DH_DIM + k] : (h16)0.f;
    }

    const h16* kbase = Kf + (size_t)(b * H_DIM + h) * (L_DIM * DH_DIM);
    float s[16][8];
#pragma unroll
    for (int nt = 0; nt < 16; ++nt) {
        int l = nt * 16 + lan15;      // score column for this lane
        if (nt < 15)   // prefetch next K tile row for this lane (global_prefetch_b8)
            __builtin_prefetch(kbase + (l + 16) * DH_DIM, 0, 3);
        v16h bt;
#pragma unroll
        for (int j = 0; j < 16; ++j) {
            int k = kmap(j, g);
            bt[j] = (k < DH_DIM) ? kbase[l * DH_DIM + k] : (h16)0.f;
        }
        v8f c = {};
        c = wmma_f16(aq, bt, c);
        unsigned int mv = mask[b * L_DIM + l];
#pragma unroll
        for (int e = 0; e < 8; ++e)
            s[nt][e] = mv ? (c[e] * 0.25f) : -1e9f;   // 1/sqrt(16)=0.25
    }

    // masked softmax over L: row m = e + 8*g lives in one 16-lane half
#pragma unroll
    for (int e = 0; e < 8; ++e) {
        float mx = -3.4e38f;
#pragma unroll
        for (int nt = 0; nt < 16; ++nt) mx = fmaxf(mx, s[nt][e]);
#pragma unroll
        for (int off = 8; off >= 1; off >>= 1) mx = fmaxf(mx, __shfl_xor(mx, off, 16));
        float sum = 0.f;
#pragma unroll
        for (int nt = 0; nt < 16; ++nt) {
            float ev = __expf(s[nt][e] - mx);
            s[nt][e] = ev;
            sum += ev;
        }
#pragma unroll
        for (int off = 8; off >= 1; off >>= 1) sum += __shfl_xor(sum, off, 16);
        float inv = 1.f / sum;
        int m = e + 8 * g;
#pragma unroll
        for (int nt = 0; nt < 16; ++nt)
            pbuf[wv][m][nt * 16 + lan15] = (h16)(s[nt][e] * inv);
    }
    __syncthreads();

    // att = exp(P @ Vlog): M=16 rows, K=L=256 (8 ksteps), N=DH=16
    // VlogT is [DH, L]: lane's B elements are two contiguous 8-half runs.
    const h16* vbase = VlogT + (size_t)(b * H_DIM + h) * (DH_DIM * L_DIM) + (size_t)lan15 * L_DIM;
    v8f acc = {};
#pragma unroll
    for (int ks = 0; ks < 8; ++ks) {
        v16h ap, bv16;
#pragma unroll
        for (int j = 0; j < 16; ++j) {
            int k = ks * 32 + kmap(j, g);        // l index
            ap[j]   = pbuf[wv][lan15][k];
            bv16[j] = vbase[k];
        }
        acc = wmma_f16(ap, bv16, acc);
    }

    const h16* rbase = rf  + (((size_t)t * B_DIM + b) * H_DIM + h) * (size_t)(V_DIM * DH_DIM);
    h16*       obase = rat + (((size_t)t * B_DIM + b) * H_DIM + h) * (size_t)(V_DIM * DH_DIM);
#pragma unroll
    for (int e = 0; e < 8; ++e) {
        int m    = e + 8 * g;
        int vrow = wv * 16 + m;
        float att = __expf(acc[e]);
        float rv  = (float)rbase[vrow * DH_DIM + lan15];
        obase[vrow * DH_DIM + lan15] = (h16)(rv * att);
    }
}

// ---------------------------------------------------------------------------
// Kernel 5: out = rat @ Wc^T + bc + vehicles - initial_pos   (f32 output)
// ---------------------------------------------------------------------------
__global__ __launch_bounds__(256) void out_kernel(
    const h16* __restrict__ rat, const float* __restrict__ Wc,
    const float* __restrict__ bc, const float* __restrict__ veh,
    const float* __restrict__ ipos, float* __restrict__ out) {
    int lane  = threadIdx.x & 31;
    int wg    = blockIdx.x * 8 + (threadIdx.x >> 5);
    int ftile = wg & 7;
    int mtile = wg >> 3;
    int g     = lane >> 4;
    int lan15 = lane & 15;
    int arow  = mtile * 16 + lan15;
    int fcol  = ftile * 16 + lan15;

    v8f c = {};
#pragma unroll
    for (int ks = 0; ks < 4; ++ks) {
        v16h a, bt;
#pragma unroll
        for (int j = 0; j < 16; ++j) {
            int k = ks * 32 + kmap(j, g);
            a[j]  = rat[(size_t)arow * F_DIM + k];
            bt[j] = (h16)Wc[fcol * F_DIM + k];
        }
        c = wmma_f16(a, bt, c);
    }
    float bias = bc[fcol];
#pragma unroll
    for (int e = 0; e < 8; ++e) {
        int m = e + 8 * g;
        size_t row = (size_t)mtile * 16 + m;
        size_t idx = row * F_DIM + fcol;
        out[idx] = c[e] + bias + veh[idx] - ipos[idx];
    }
}

// ---------------------------------------------------------------------------
extern "C" void kernel_launch(void* const* d_in, const int* in_sizes, int n_in,
                              void* d_out, int out_size, void* d_ws, size_t ws_size,
                              hipStream_t stream) {
    const float* vehicles = (const float*)d_in[0];
    const float* ipos     = (const float*)d_in[1];
    const float* lanes    = (const float*)d_in[2];
    const unsigned char* mask_lanes = (const unsigned char*)d_in[3];  // numpy bool = 1 byte
    const float* Wk = (const float*)d_in[4];
    const float* bk = (const float*)d_in[5];
    const float* Wv = (const float*)d_in[6];
    const float* bv = (const float*)d_in[7];
    const float* Wq = (const float*)d_in[8];
    const float* bq = (const float*)d_in[9];
    const float* Wr = (const float*)d_in[10];
    const float* br = (const float*)d_in[11];
    const float* Wc = (const float*)d_in[12];
    const float* bc = (const float*)d_in[13];
    float* out = (float*)d_out;

    // workspace carve-up (all fully re-written every call)
    char* ws = (char*)d_ws;
    size_t off = 0;
    auto carve = [&](size_t bytes) -> char* {
        char* p = ws + off;
        off += (bytes + 255) & ~(size_t)255;
        return p;
    };
    const size_t laneElems = (size_t)B_DIM * H_DIM * L_DIM * DH_DIM;           // 1,048,576
    const size_t vehElems  = (size_t)T_DIM * B_DIM * H_DIM * V_DIM * DH_DIM;   // 3,932,160
    h16* Kf    = (h16*)carve(laneElems * sizeof(h16));
    h16* VlogT = (h16*)carve(laneElems * sizeof(h16));
    unsigned int* maskp = (unsigned int*)carve((size_t)B_DIM * L_DIM * sizeof(unsigned int));
    h16* qf    = (h16*)carve(vehElems * sizeof(h16));
    h16* rf    = (h16*)carve(vehElems * sizeof(h16));
    h16* rat   = (h16*)carve(vehElems * sizeof(h16));
    (void)ws_size; (void)in_sizes; (void)n_in; (void)out_size;

    lane_proj_kernel<<<dim3(512),  dim3(256), 0, stream>>>(lanes, Wk, bk, Wv, bv, Kf, VlogT);
    veh_proj_kernel <<<dim3(1920), dim3(256), 0, stream>>>(vehicles, ipos, Wq, bq, Wr, br, qf, rf);
    mask_kernel     <<<dim3(32),   dim3(256), 0, stream>>>(mask_lanes, maskp);
    attn_kernel     <<<dim3(7680), dim3(64),  0, stream>>>(qf, Kf, VlogT, maskp, rf, rat);
    out_kernel      <<<dim3(1920), dim3(256), 0, stream>>>(rat, Wc, bc, vehicles, ipos, out);
}
